// LightningIndexer_48524540510839
// MI455X (gfx1250) — matrix operand order repaired and checked by reading, not compiled, not run on hardware
//
#include <hip/hip_runtime.h>

#define TQ    1024
#define TKD   4096
#define CDIM  2048
#define NH    32
#define HD    64
#define HDTOT 2048
#define NTOPK 2048

typedef __attribute__((ext_vector_type(16))) __bf16 v16bf;
typedef __attribute__((ext_vector_type(8)))  float  v8f;

// LDS byte-offset of a __shared__ pointer: flat LDS aperture keeps the
// offset in addr[31:0] (CDNA5 ISA 10.2 aperture table).
__device__ __forceinline__ unsigned lds_off(const void* p) {
  return (unsigned)(unsigned long long)p;
}

// CDNA5 async copy global->LDS, 16B per lane, tracked by ASYNCcnt.
__device__ __forceinline__ void async_copy_b128(unsigned lds_byte_off,
                                                const void* gaddr) {
  asm volatile("global_load_async_to_lds_b128 %0, %1, off"
               :: "v"(lds_byte_off), "v"(gaddr)
               : "memory");
}
__device__ __forceinline__ void wait_asynccnt0() {
  asm volatile("s_wait_asynccnt 0x0" ::: "memory");
}

// ---------------------------------------------------------------------------
// Projection: Out[M][2048] (bf16) = X[M][C] @ W[2048][C]^T
// grid = (2048/128, M/128), block = 256 (8 waves), 128x128 tile, K step 32.
// Each wave owns 2 (M) x 4 (N) subtiles -> 8 WMMAs per K-step, A/B frags
// reused 4x/2x. f32 inputs converted to bf16 through VGPRs (v_cvt_pk_bf16).
// ---------------------------------------------------------------------------
__global__ __launch_bounds__(256) void li_proj_kernel(
    const float* __restrict__ X, const float* __restrict__ W,
    __bf16* __restrict__ Out)
{
  constexpr int PS = 40;                       // padded LDS row stride (elems)
  __shared__ __bf16 As[128 * PS];              // X tile  (rows m, cols c)
  __shared__ __bf16 Bs[128 * PS];              // W tile  (rows n, cols c)

  const int bm   = blockIdx.y * 128;
  const int bn   = blockIdx.x * 128;
  const int tid  = threadIdx.x;
  const int lane = tid & 31;
  const int wave = tid >> 5;
  const int half = lane >> 4;
  const int lrow = lane & 15;
  const int mg   = wave & 3;                   // M subtiles mg*2 + {0,1}
  const int ngp  = wave >> 2;                  // N subtiles ngp*4 + {0..3}

  v8f acc[2][4];
#pragma unroll
  for (int mi = 0; mi < 2; ++mi)
#pragma unroll
    for (int ni = 0; ni < 4; ++ni) acc[mi][ni] = (v8f){};

  for (int kc = 0; kc < CDIM; kc += 32) {
    // stage 128x32 f32 -> bf16 for both operands (1024 float4 each)
    for (int v = tid; v < 1024; v += 256) {
      const int r  = v >> 3;
      const int c4 = (v & 7) * 4;
      float4 xv = *(const float4*)(X + (size_t)(bm + r) * CDIM + kc + c4);
      As[r * PS + c4 + 0] = (__bf16)xv.x;
      As[r * PS + c4 + 1] = (__bf16)xv.y;
      As[r * PS + c4 + 2] = (__bf16)xv.z;
      As[r * PS + c4 + 3] = (__bf16)xv.w;
      float4 wv = *(const float4*)(W + (size_t)(bn + r) * CDIM + kc + c4);
      Bs[r * PS + c4 + 0] = (__bf16)wv.x;
      Bs[r * PS + c4 + 1] = (__bf16)wv.y;
      Bs[r * PS + c4 + 2] = (__bf16)wv.z;
      Bs[r * PS + c4 + 3] = (__bf16)wv.w;
    }
    __syncthreads();

    // A fragments (16x32 bf16): K = half*16 + j
    v16bf a[2];
#pragma unroll
    for (int mi = 0; mi < 2; ++mi)
#pragma unroll
      for (int j = 0; j < 16; ++j) {
        const int kk = half * 16 + j;
        a[mi][j] = As[((mg * 2 + mi) * 16 + lrow) * PS + kk];
      }
    // B fragments (32x16 bf16): lane = N, K = half*16 + j
    v16bf b[4];
#pragma unroll
    for (int ni = 0; ni < 4; ++ni)
#pragma unroll
      for (int j = 0; j < 16; ++j) {
        const int kk = half * 16 + j;
        b[ni][j] = Bs[((ngp * 4 + ni) * 16 + lrow) * PS + kk];
      }
#pragma unroll
    for (int mi = 0; mi < 2; ++mi)
#pragma unroll
      for (int ni = 0; ni < 4; ++ni)
        acc[mi][ni] = __builtin_amdgcn_wmma_f32_16x16x32_bf16(
            false, a[mi], false, b[ni], (short)0, acc[mi][ni], false, false);
    __syncthreads();
  }

  // store bf16 result, C/D layout: M = vgpr + 8*half, N = lane&15
#pragma unroll
  for (int mi = 0; mi < 2; ++mi)
#pragma unroll
    for (int ni = 0; ni < 4; ++ni)
#pragma unroll
      for (int j = 0; j < 8; ++j) {
        const int m = bm + (mg * 2 + mi) * 16 + (half ? 8 : 0) + j;
        const int n = bn + (ngp * 4 + ni) * 16 + lrow;
        Out[(size_t)m * HDTOT + n] = (__bf16)acc[mi][ni][j];
      }
}

// ---------------------------------------------------------------------------
// Scores: scores[q][k] = sum_h softmax(hw)[h] * relu(scale * <iq[q,h,:], ik[k,h,:]>)
// grid = (4096/128, 1024/64), block = 256 (8 waves).
// 64(q) x 128(k) tile; per wave 4 subtiles (A-frag reused 4x).
// Tiles are bf16 in global already -> staged with global_load_async_to_lds_b128.
// ---------------------------------------------------------------------------
__global__ __launch_bounds__(256) void li_scores_kernel(
    const __bf16* __restrict__ iq, const __bf16* __restrict__ ik,
    const float* __restrict__ hw, float* __restrict__ scores)
{
  constexpr int PS = 72;                       // 144B rows: 16B-aligned, odd banks
  __shared__ __bf16 Qs[64 * PS];               //  ~9.2 KB
  __shared__ __bf16 Ks[128 * PS];              // ~18.4 KB
  __shared__ float  wsh[NH];
  __shared__ float  wsum;

  const int tid = threadIdx.x;
  if (tid < NH) wsh[tid] = __expf(hw[tid]);
  __syncthreads();
  if (tid == 0) { float s = 0.f; for (int i = 0; i < NH; ++i) s += wsh[i]; wsum = s; }
  __syncthreads();
  const float wscale = 0.125f / wsum;          // 1/sqrt(64) folded into weight

  const int bq   = blockIdx.y * 64;
  const int bk   = blockIdx.x * 128;
  const int lane = tid & 31;
  const int wave = tid >> 5;
  const int half = lane >> 4;
  const int lrow = lane & 15;
  const int mt   = wave & 3;                   // M subtile 0..3
  const int ng   = (wave >> 2) * 4;            // first of 4 N subtiles (0 or 4)

  v8f s0 = {}, s1 = {}, s2 = {}, s3 = {};

  const unsigned qbase = lds_off(Qs);
  const unsigned kbase = lds_off(Ks);

  for (int h = 0; h < NH; ++h) {
    // ---- async stage: Q tile 64x64 (512 x 16B), K tile 128x64 (1024 x 16B)
    for (int v = tid; v < 512; v += 256) {
      const int r  = v >> 3;
      const int c8 = (v & 7) * 8;
      async_copy_b128(qbase + (unsigned)(r * PS + c8) * 2,
                      iq + (size_t)(bq + r) * HDTOT + h * HD + c8);
    }
    for (int v = tid; v < 1024; v += 256) {
      const int r  = v >> 3;
      const int c8 = (v & 7) * 8;
      async_copy_b128(kbase + (unsigned)(r * PS + c8) * 2,
                      ik + (size_t)(bk + r) * HDTOT + h * HD + c8);
    }
    if (h + 1 < NH) {                          // global_prefetch_b8 next head
      __builtin_prefetch(iq + (size_t)(bq + (tid & 63)) * HDTOT + (h + 1) * HD, 0, 0);
      __builtin_prefetch(ik + (size_t)(bk + (tid & 127)) * HDTOT + (h + 1) * HD, 0, 0);
    }
    wait_asynccnt0();
    __syncthreads();

    v8f h0 = {}, h1 = {}, h2 = {}, h3 = {};    // per-head accumulators
#pragma unroll
    for (int ks = 0; ks < HD; ks += 32) {
      v16bf a;
#pragma unroll
      for (int j = 0; j < 16; ++j) {
        const int kk = ks + half * 16 + j;
        a[j] = Qs[(mt * 16 + lrow) * PS + kk];
      }
      v16bf b0, b1, b2, b3;
#pragma unroll
      for (int j = 0; j < 16; ++j) {
        const int kk = ks + half * 16 + j;
        b0[j] = Ks[((ng + 0) * 16 + lrow) * PS + kk];
        b1[j] = Ks[((ng + 1) * 16 + lrow) * PS + kk];
        b2[j] = Ks[((ng + 2) * 16 + lrow) * PS + kk];
        b3[j] = Ks[((ng + 3) * 16 + lrow) * PS + kk];
      }
      h0 = __builtin_amdgcn_wmma_f32_16x16x32_bf16(false, a, false, b0,
                                                   (short)0, h0, false, false);
      h1 = __builtin_amdgcn_wmma_f32_16x16x32_bf16(false, a, false, b1,
                                                   (short)0, h1, false, false);
      h2 = __builtin_amdgcn_wmma_f32_16x16x32_bf16(false, a, false, b2,
                                                   (short)0, h2, false, false);
      h3 = __builtin_amdgcn_wmma_f32_16x16x32_bf16(false, a, false, b3,
                                                   (short)0, h3, false, false);
    }
    const float w = wsh[h] * wscale;
#pragma unroll
    for (int j = 0; j < 8; ++j) {
      s0[j] += w * fmaxf(h0[j], 0.f);
      s1[j] += w * fmaxf(h1[j], 0.f);
      s2[j] += w * fmaxf(h2[j], 0.f);
      s3[j] += w * fmaxf(h3[j], 0.f);
    }
    __syncthreads();
  }

#pragma unroll
  for (int j = 0; j < 8; ++j) {
    const int q = bq + mt * 16 + (half ? 8 : 0) + j;
    float* rowp = scores + (size_t)q * TKD + bk;
    rowp[(ng + 0) * 16 + lrow] = s0[j];
    rowp[(ng + 1) * 16 + lrow] = s1[j];
    rowp[(ng + 2) * 16 + lrow] = s2[j];
    rowp[(ng + 3) * 16 + lrow] = s3[j];
  }
}

// ---------------------------------------------------------------------------
// Top-k: one block per q row. scores >= 0 (relu x positive softmax weights),
// so pack (score_bits<<32)|~idx and bitonic-sort u64 descending; ~idx makes
// lower index win ties, matching jax.lax.top_k. Emit first 2048.
// ---------------------------------------------------------------------------
__global__ __launch_bounds__(256) void li_topk_kernel(
    const float* __restrict__ scores, int* __restrict__ out_idx,
    float* __restrict__ out_scr)
{
  __shared__ unsigned long long keys[TKD];     // 32 KB
  const int q = blockIdx.x;
  const float* row = scores + (size_t)q * TKD;

  for (int i = threadIdx.x; i < TKD; i += 256) {
    const unsigned int b = __float_as_uint(row[i]);
    keys[i] = ((unsigned long long)b << 32) | (unsigned int)(~i);
  }
  __syncthreads();

  for (int k = 2; k <= TKD; k <<= 1) {
    for (int j = k >> 1; j > 0; j >>= 1) {
      for (int i = threadIdx.x; i < TKD; i += 256) {
        const int l = i ^ j;
        if (l > i) {
          const unsigned long long a = keys[i];
          const unsigned long long c = keys[l];
          const bool desc = ((i & k) == 0);
          if (desc ? (a < c) : (a > c)) { keys[i] = c; keys[l] = a; }
        }
      }
      __syncthreads();
    }
  }

  for (int i = threadIdx.x; i < NTOPK; i += 256) {
    const unsigned long long kv = keys[i];
    out_idx[(size_t)q * NTOPK + i] = (int)(~(unsigned int)kv);
    out_scr[(size_t)q * NTOPK + i] = __uint_as_float((unsigned int)(kv >> 32));
  }
}

// ---------------------------------------------------------------------------
extern "C" void kernel_launch(void* const* d_in, const int* in_sizes, int n_in,
                              void* d_out, int out_size, void* d_ws, size_t ws_size,
                              hipStream_t stream)
{
  const float* query = (const float*)d_in[0];   // [1,1024,2048]
  const float* key   = (const float*)d_in[1];   // [1,4096,2048]
  const float* Wq    = (const float*)d_in[2];   // [2048,2048]
  const float* Wk    = (const float*)d_in[3];   // [2048,2048]
  const float* hw    = (const float*)d_in[4];   // [32]

  char* ws = (char*)d_ws;
  __bf16* iq = (__bf16*)ws;                                    //  4 MB
  __bf16* ik = (__bf16*)(ws + (size_t)TQ * HDTOT * 2);         // 16 MB
  float*  sc = (float*)(ws + (size_t)TQ * HDTOT * 2
                           + (size_t)TKD * HDTOT * 2);         // 16 MB

  int*   out_idx = (int*)d_out;                                // [1024,2048] i32
  float* out_scr = (float*)d_out + (size_t)TQ * NTOPK;         // [1024,2048] f32

  li_proj_kernel<<<dim3(HDTOT / 128, TQ / 128), 256, 0, stream>>>(query, Wq, iq);
  li_proj_kernel<<<dim3(HDTOT / 128, TKD / 128), 256, 0, stream>>>(key, Wk, ik);
  li_scores_kernel<<<dim3(TKD / 128, TQ / 64), 256, 0, stream>>>(iq, ik, hw, sc);
  li_topk_kernel<<<dim3(TQ), 256, 0, stream>>>(sc, out_idx, out_scr);
}
